// MoE_25443386262322
// MI455X (gfx1250) — compile-verified, matrix operand order checked by reading, not run on hardware
//
#include <hip/hip_runtime.h>
#include <hip/hip_bf16.h>

// ---------------------------------------------------------------------------
// MoE (DeepSeek-style) for MI455X / gfx1250: sparse grouped-GEMM in bf16 WMMA.
// Workspace layout (~68 MB assumed available in d_ws):
//   combine[T*E] f32 | counts[32] | offsets[32] | cursor[32] | toks[T*TOPK]
//   hbuf[T*TOPK*INTER] bf16 | shbuf[T*SHARED_INTER] bf16
// ---------------------------------------------------------------------------

#define T_TOK 4096
#define DIM   2048
#define INTER 1024
#define NEXP  32
#define TOPK  6
#define NGRP  8
#define TOPKG 4
#define SHINTER 2048
#define ROUTE_SCALE 1.0f

typedef __attribute__((ext_vector_type(16))) __bf16 v16bf;
typedef __attribute__((ext_vector_type(2)))  __bf16 v2bf;
typedef __attribute__((ext_vector_type(2)))  float  v2f;
typedef __attribute__((ext_vector_type(8)))  float  v8f;

union FragU {
    uint4 u[2];
    v16bf v;
};

// fp32 -> bf16 via native fptrunc (lowers to v_cvt_*_bf16 on gfx1250).
__device__ __forceinline__ unsigned short f2bf(float f) {
    union { __bf16 b; unsigned short s; } u;
    u.b = (__bf16)f;
    return u.s;
}
// adjacent pair -> one v_cvt_pk_bf16_f32, no repacking
__device__ __forceinline__ unsigned f2bf2(float lo, float hi) {
    v2f a = {lo, hi};
    union { v2bf b; unsigned u; } p;
    p.b = __builtin_convertvector(a, v2bf);
    return p.u;
}

#if __has_builtin(__builtin_amdgcn_global_load_async_to_lds_b128)
#define HAVE_ASYNC_LDS 1
typedef __attribute__((ext_vector_type(4))) int v4i;
typedef __attribute__((address_space(1))) v4i* gv4i_p;   // global (printed "__device__")
typedef __attribute__((address_space(3))) v4i* lv4i_p;   // LDS
#else
#define HAVE_ASYNC_LDS 0
#endif

// ------------------------------- gate ---------------------------------------
__global__ void moe_init_counts(int* counts) {
    if (threadIdx.x < NEXP) counts[threadIdx.x] = 0;
}

__global__ void moe_gate_kernel(const float* __restrict__ x,
                                const float* __restrict__ gw,
                                float* __restrict__ combine,
                                int* __restrict__ counts) {
    const int t = blockIdx.x;
    __shared__ float xs[DIM];
    __shared__ float sc[NEXP];
    const float* xr = x + (size_t)t * DIM;
    for (int k = threadIdx.x; k < DIM; k += 256) xs[k] = xr[k];
    __syncthreads();

    const int wv = threadIdx.x >> 5;
    const int ln = threadIdx.x & 31;
    for (int i = 0; i < 4; ++i) {
        const int e = wv * 4 + i;
        const float* wr = gw + (size_t)e * DIM;
        float s = 0.f;
        for (int k = ln; k < DIM; k += 32) s += xs[k] * wr[k];
        for (int o = 16; o > 0; o >>= 1) s += __shfl_down(s, o, 32);
        if (ln == 0) sc[e] = s;
    }
    __syncthreads();

    if (threadIdx.x == 0) {
        float v[NEXP];
        float mx = -1e30f;
        for (int i = 0; i < NEXP; ++i) { v[i] = sc[i]; mx = fmaxf(mx, v[i]); }
        float sum = 0.f;
        for (int i = 0; i < NEXP; ++i) { v[i] = __expf(v[i] - mx); sum += v[i]; }
        const float inv = 1.f / sum;
        for (int i = 0; i < NEXP; ++i) v[i] *= inv;

        float g[NGRP];
        for (int gi = 0; gi < NGRP; ++gi) {
            float m = v[gi * 4];
            for (int j = 1; j < 4; ++j) m = fmaxf(m, v[gi * 4 + j]);
            g[gi] = m;
        }
        bool gsel[NGRP];
        for (int gi = 0; gi < NGRP; ++gi) gsel[gi] = false;
        for (int r = 0; r < TOPKG; ++r) {        // top-4 groups, lowest index on tie
            int best = -1; float bv = -1e30f;
            for (int gi = 0; gi < NGRP; ++gi)
                if (!gsel[gi] && g[gi] > bv) { bv = g[gi]; best = gi; }
            gsel[best] = true;
        }
        float cmb[NEXP];
        bool esel[NEXP];
        for (int i = 0; i < NEXP; ++i) { cmb[i] = 0.f; esel[i] = false; }
        for (int r = 0; r < TOPK; ++r) {          // top-6 experts within allowed groups
            int best = -1; float bv = -1e30f;
            for (int i = 0; i < NEXP; ++i)
                if (gsel[i >> 2] && !esel[i] && v[i] > bv) { bv = v[i]; best = i; }
            esel[best] = true;
            cmb[best] = v[best] * ROUTE_SCALE;
            atomicAdd(&counts[best], 1);
        }
        float* crow = combine + (size_t)t * NEXP;
        for (int i = 0; i < NEXP; ++i) crow[i] = cmb[i];
    }
}

__global__ void moe_scan_kernel(const int* __restrict__ counts,
                                int* __restrict__ offsets,
                                int* __restrict__ cursor) {
    if (threadIdx.x == 0) {
        int a = 0;
        for (int e = 0; e < NEXP; ++e) { offsets[e] = a; a += counts[e]; }
    }
    if (threadIdx.x < NEXP) cursor[threadIdx.x] = 0;
}

__global__ void moe_scatter_kernel(const float* __restrict__ combine,
                                   const int* __restrict__ offsets,
                                   int* __restrict__ cursor,
                                   int* __restrict__ toks) {
    const int t = blockIdx.x * blockDim.x + threadIdx.x;
    if (t >= T_TOK) return;
    const float* crow = combine + (size_t)t * NEXP;
    for (int e = 0; e < NEXP; ++e)
        if (crow[e] > 0.f) {
            const int s = atomicAdd(&cursor[e], 1);
            toks[offsets[e] + s] = t;
        }
}

// --------------------------- WMMA GEMM stages -------------------------------
#define TM 128
#define TN 64
#define TK 32
#define LDA 40          // TK + 8 pad: 80-byte LDS rows, 16B-aligned

// Stage 1: H = silu(A@Wg) * (A@Wu), A gathered fp32 rows -> bf16 output.
template <bool ROUTED>
__global__ void __launch_bounds__(256)
ffn_stage1(const float* __restrict__ X,
           const float* __restrict__ WgAll, const float* __restrict__ WuAll,
           const int* __restrict__ counts, const int* __restrict__ offsets,
           const int* __restrict__ toks,
           unsigned short* __restrict__ H, int N) {
    const int e = ROUTED ? blockIdx.z : 0;
    const int cnt = ROUTED ? counts[e] : T_TOK;
    const int rowbase = blockIdx.y * TM;
    if (rowbase >= cnt) return;
    const float* Wg = WgAll + (size_t)e * DIM * N;
    const float* Wu = WuAll + (size_t)e * DIM * N;
    const int n0 = blockIdx.x * TN;

    __shared__ __align__(16) unsigned short As[TM * LDA];
    __shared__ __align__(16) unsigned short Bgs[TN * LDA];
    __shared__ __align__(16) unsigned short Bus[TN * LDA];
    __shared__ int tok_s[TM];

    const int tid = threadIdx.x;
    if (tid < TM) {
        const int r = rowbase + tid;
        if constexpr (ROUTED)
            tok_s[tid] = (r < cnt) ? toks[offsets[e] + r] : toks[offsets[e]];
        else
            tok_s[tid] = (r < T_TOK) ? r : 0;
    }
    __syncthreads();

    // hoist loop-invariant gather pointers / LDS destinations out of the k-loop
    const float* arow[4];
    unsigned short* adst[4];
    {
#pragma unroll
        for (int j = 0; j < 4; ++j) {
            const int id = tid + j * 256;
            const int m = id >> 3;
            const int c4 = (id & 7) * 4;
            arow[j] = X + (size_t)tok_s[m] * DIM + c4;
            adst[j] = &As[m * LDA + c4];
        }
    }
    const int nl = tid & 63;
    const int kl0 = (tid >> 6) * 8;
    const float* gp0 = Wg + (size_t)kl0 * N + n0 + nl;
    const float* up0 = Wu + (size_t)kl0 * N + n0 + nl;
    unsigned short* bgd = &Bgs[nl * LDA + kl0];
    unsigned short* bud = &Bus[nl * LDA + kl0];

    const int ln = tid & 31, wv = tid >> 5;
    const int wm = wv >> 1, wn = wv & 1;
    const int lr = ln & 15;
    const int kb = (ln < 16) ? 0 : 8;

    v8f zero = {0.f, 0.f, 0.f, 0.f, 0.f, 0.f, 0.f, 0.f};
    v8f accg[2][2], accu[2][2];
    for (int i = 0; i < 2; ++i)
        for (int j = 0; j < 2; ++j) { accg[i][j] = zero; accu[i][j] = zero; }

    for (int k0 = 0; k0 < DIM; k0 += TK) {
        // ---- issue ALL global loads first (clauses + partial loadcnt waits) ----
        float4 va[4];
#pragma unroll
        for (int j = 0; j < 4; ++j) va[j] = *(const float4*)(arow[j] + k0);
        float bgv[8], buv[8];
        {
            const float* gp = gp0 + (size_t)k0 * N;
            const float* up = up0 + (size_t)k0 * N;
#pragma unroll
            for (int i = 0; i < 8; ++i) {
                bgv[i] = gp[(size_t)i * N];
                buv[i] = up[(size_t)i * N];
            }
            if (k0 + TK < DIM) {
                __builtin_prefetch(gp + (size_t)TK * N, 0, 1);
                __builtin_prefetch(up + (size_t)TK * N, 0, 1);
            }
        }
        // ---- convert + stage to LDS ----
#pragma unroll
        for (int j = 0; j < 4; ++j) {
            uint2 p;
            p.x = f2bf2(va[j].x, va[j].y);
            p.y = f2bf2(va[j].z, va[j].w);
            *(uint2*)adst[j] = p;
        }
#pragma unroll
        for (int i = 0; i < 8; i += 2) {
            *(unsigned*)(bgd + i) = f2bf2(bgv[i], bgv[i + 1]);
            *(unsigned*)(bud + i) = f2bf2(buv[i], buv[i + 1]);
        }
        __syncthreads();

        FragU a[2], bg[2], bu[2];
#pragma unroll
        for (int s = 0; s < 2; ++s) {
            const int m = wm * 32 + s * 16 + lr;
            a[s].u[0] = *(const uint4*)&As[m * LDA + kb];
            a[s].u[1] = *(const uint4*)&As[m * LDA + kb + 16];
            const int n = wn * 32 + s * 16 + lr;
            bg[s].u[0] = *(const uint4*)&Bgs[n * LDA + kb];
            bg[s].u[1] = *(const uint4*)&Bgs[n * LDA + kb + 16];
            bu[s].u[0] = *(const uint4*)&Bus[n * LDA + kb];
            bu[s].u[1] = *(const uint4*)&Bus[n * LDA + kb + 16];
        }
#pragma unroll
        for (int i = 0; i < 2; ++i)
#pragma unroll
            for (int j = 0; j < 2; ++j) {
                accg[i][j] = __builtin_amdgcn_wmma_f32_16x16x32_bf16(
                    false, a[i].v, false, bg[j].v, (short)0, accg[i][j], false, false);
                accu[i][j] = __builtin_amdgcn_wmma_f32_16x16x32_bf16(
                    false, a[i].v, false, bu[j].v, (short)0, accu[i][j], false, false);
            }
        __syncthreads();
    }

    // epilogue: h = silu(g)*u -> bf16
    const int hi = (ln >> 4) << 3;   // +8 rows for lanes 16-31
#pragma unroll
    for (int i = 0; i < 2; ++i)
#pragma unroll
        for (int j = 0; j < 2; ++j) {
            const int nloc = n0 + wn * 32 + j * 16 + lr;
#pragma unroll
            for (int r = 0; r < 8; ++r) {
                const int mloc = wm * 32 + i * 16 + r + hi;
                const int row = rowbase + mloc;
                if (row < cnt) {
                    const float hg = accg[i][j][r];
                    const float hu = accu[i][j][r];
                    const float h = (hg / (1.f + __expf(-hg))) * hu;
                    const size_t slot = ROUTED ? (size_t)(offsets[e] + row) : (size_t)row;
                    H[slot * (size_t)N + nloc] = f2bf(h);
                }
            }
        }
}

// Stage 2: Y = H @ Wd (scaled by routing weight, atomic add) or direct store.
template <bool ROUTED>
__global__ void __launch_bounds__(256)
ffn_stage2(const unsigned short* __restrict__ H,
           const float* __restrict__ WdAll,
           const int* __restrict__ counts, const int* __restrict__ offsets,
           const int* __restrict__ toks, const float* __restrict__ combine,
           float* __restrict__ out, int K) {
    const int e = ROUTED ? blockIdx.z : 0;
    const int cnt = ROUTED ? counts[e] : T_TOK;
    const int rowbase = blockIdx.y * TM;
    if (rowbase >= cnt) return;
    const float* Wd = WdAll + (size_t)e * K * DIM;
    const int n0 = blockIdx.x * TN;
    const size_t hbase = ROUTED ? (size_t)offsets[e] : 0;

    __shared__ __align__(16) unsigned short As[TM * LDA];
    __shared__ __align__(16) unsigned short Bs[TN * LDA];
    __shared__ int tok_s[TM];

    const int tid = threadIdx.x;
    if (tid < TM) {
        const int r = rowbase + tid;
        if constexpr (ROUTED)
            tok_s[tid] = (r < cnt) ? toks[offsets[e] + r] : 0;
        else
            tok_s[tid] = r;
    }
    __syncthreads();

    // hoisted per-thread A-tile source rows (bf16) and LDS destinations
    const unsigned short* hrow[2];
    unsigned short* adst[2];
#pragma unroll
    for (int j = 0; j < 2; ++j) {
        const int id = tid + j * 256;
        const int m = id >> 2;
        const int c = (id & 3) * 8;
        int r = rowbase + m;
        if (r >= cnt) r = cnt - 1;              // clamp (never stored)
        hrow[j] = H + (hbase + r) * (size_t)K + c;
        adst[j] = &As[m * LDA + c];
    }
    const int nl = tid & 63;
    const int kl0 = (tid >> 6) * 8;
    const float* dp0 = Wd + (size_t)kl0 * DIM + n0 + nl;
    unsigned short* bdst = &Bs[nl * LDA + kl0];

    const int ln = tid & 31, wv = tid >> 5;
    const int wm = wv >> 1, wn = wv & 1;
    const int lr = ln & 15;
    const int kb = (ln < 16) ? 0 : 8;

    v8f zero = {0.f, 0.f, 0.f, 0.f, 0.f, 0.f, 0.f, 0.f};
    v8f acc[2][2];
    for (int i = 0; i < 2; ++i)
        for (int j = 0; j < 2; ++j) acc[i][j] = zero;

    for (int k0 = 0; k0 < K; k0 += TK) {
        // A tile: bf16 rows, 16B per thread-chunk; async DMA to LDS when available
#if HAVE_ASYNC_LDS
#pragma unroll
        for (int j = 0; j < 2; ++j)
            __builtin_amdgcn_global_load_async_to_lds_b128(
                (gv4i_p)(hrow[j] + k0), (lv4i_p)adst[j], 0, 0);
#else
#pragma unroll
        for (int j = 0; j < 2; ++j)
            *(uint4*)adst[j] = *(const uint4*)(hrow[j] + k0);
#endif
        // B tile: issue all loads first, then packed fp32->bf16 + LDS store
        {
            const float* dp = dp0 + (size_t)k0 * DIM;
            float bv[8];
#pragma unroll
            for (int i = 0; i < 8; ++i) bv[i] = dp[(size_t)i * DIM];
            if (k0 + TK < K) __builtin_prefetch(dp + (size_t)TK * DIM, 0, 1);
#pragma unroll
            for (int i = 0; i < 8; i += 2)
                *(unsigned*)(bdst + i) = f2bf2(bv[i], bv[i + 1]);
        }
#if HAVE_ASYNC_LDS
        asm volatile("s_wait_asynccnt 0x0" ::: "memory");
#endif
        __syncthreads();

        FragU a[2], b[2];
#pragma unroll
        for (int s = 0; s < 2; ++s) {
            const int m = wm * 32 + s * 16 + lr;
            a[s].u[0] = *(const uint4*)&As[m * LDA + kb];
            a[s].u[1] = *(const uint4*)&As[m * LDA + kb + 16];
            const int n = wn * 32 + s * 16 + lr;
            b[s].u[0] = *(const uint4*)&Bs[n * LDA + kb];
            b[s].u[1] = *(const uint4*)&Bs[n * LDA + kb + 16];
        }
#pragma unroll
        for (int i = 0; i < 2; ++i)
#pragma unroll
            for (int j = 0; j < 2; ++j)
                acc[i][j] = __builtin_amdgcn_wmma_f32_16x16x32_bf16(
                    false, a[i].v, false, b[j].v, (short)0, acc[i][j], false, false);
        __syncthreads();
    }

    const int hi = (ln >> 4) << 3;
#pragma unroll
    for (int i = 0; i < 2; ++i)
#pragma unroll
        for (int j = 0; j < 2; ++j) {
            const int nloc = n0 + wn * 32 + j * 16 + lr;
#pragma unroll
            for (int r = 0; r < 8; ++r) {
                const int mloc = wm * 32 + i * 16 + r + hi;
                const int row = rowbase + mloc;
                if (row < cnt) {
                    const float v = acc[i][j][r];
                    if constexpr (ROUTED) {
                        const int t = tok_s[mloc];
                        const float w = combine[(size_t)t * NEXP + e];
                        atomicAdd(&out[(size_t)t * DIM + nloc], v * w);
                    } else {
                        out[(size_t)row * DIM + nloc] = v;   // overwrites poison
                    }
                }
            }
        }
}

// ------------------------------- launch -------------------------------------
extern "C" void kernel_launch(void* const* d_in, const int* in_sizes, int n_in,
                              void* d_out, int out_size, void* d_ws, size_t ws_size,
                              hipStream_t stream) {
    (void)in_sizes; (void)n_in; (void)out_size; (void)ws_size;
    const float* x      = (const float*)d_in[0];
    const float* gate_w = (const float*)d_in[1];
    const float* wg     = (const float*)d_in[2];
    const float* wu     = (const float*)d_in[3];
    const float* wd     = (const float*)d_in[4];
    const float* swg    = (const float*)d_in[5];
    const float* swu    = (const float*)d_in[6];
    const float* swd    = (const float*)d_in[7];
    float* out = (float*)d_out;

    char* ws = (char*)d_ws;
    float* combine = (float*)ws;                       ws += (size_t)T_TOK * NEXP * sizeof(float);
    int* counts  = (int*)ws;                           ws += 128;
    int* offsets = (int*)ws;                           ws += 128;
    int* cursor  = (int*)ws;                           ws += 128;
    int* toks    = (int*)ws;                           ws += ((size_t)T_TOK * TOPK * 4 + 255) & ~(size_t)255;
    unsigned short* hbuf  = (unsigned short*)ws;       ws += (size_t)T_TOK * TOPK * INTER * 2;
    unsigned short* shbuf = (unsigned short*)ws;

    moe_init_counts<<<1, 64, 0, stream>>>(counts);
    moe_gate_kernel<<<T_TOK, 256, 0, stream>>>(x, gate_w, combine, counts);
    moe_scan_kernel<<<1, 64, 0, stream>>>(counts, offsets, cursor);
    moe_scatter_kernel<<<T_TOK / 256, 256, 0, stream>>>(combine, offsets, cursor, toks);

    // Shared experts: dense MLP; stage-2 overwrites d_out (must precede routed adds).
    dim3 g1s(SHINTER / TN, T_TOK / TM, 1);
    ffn_stage1<false><<<g1s, 256, 0, stream>>>(x, swg, swu, counts, offsets, toks, shbuf, SHINTER);
    dim3 g2s(DIM / TN, T_TOK / TM, 1);
    ffn_stage2<false><<<g2s, 256, 0, stream>>>(shbuf, swd, counts, offsets, toks, combine, out, SHINTER);

    // Routed experts: grouped GEMM over gathered token lists; early-exit tiles.
    dim3 g1r(INTER / TN, T_TOK / TM, NEXP);
    ffn_stage1<true><<<g1r, 256, 0, stream>>>(x, wg, wu, counts, offsets, toks, hbuf, INTER);
    dim3 g2r(DIM / TN, T_TOK / TM, NEXP);
    ffn_stage2<true><<<g2r, 256, 0, stream>>>(hbuf, wd, counts, offsets, toks, combine, out, INTER);
}
